// Attention_11897059410543
// MI455X (gfx1250) — matrix operand order repaired and checked by reading, not compiled, not run on hardware
//
#include <hip/hip_runtime.h>
#include <hip/hip_bf16.h>

typedef __attribute__((ext_vector_type(16))) _Float16 v16h;
typedef __attribute__((ext_vector_type(8)))  float    v8f;
typedef __attribute__((ext_vector_type(4)))  float    f32x4;  // native vec4

// Problem constants from the reference: B=16, S=2048, D=64.
constexpr int BATCH = 16;
constexpr int SEQ   = 2048;
constexpr int DIM   = 64;
constexpr int LDP   = SEQ + 4;       // padded LDS row stride: rows shift 4 banks
constexpr float SCALE    = 0.125f;   // 1/sqrt(64)
constexpr float MASK_NEG = -1e9f;

// 16-bit A-operand K index for VGPR-half v (0..7), half-wave h (0/1):
//   v<4 -> K base 0, v>=4 -> K base 16 ; +8 per half-wave ; pairs of 2.
__device__ __forceinline__ int kmapA(int v, int h) {
    return ((v & 4) << 2) + (h << 3) + ((v & 3) << 1);
}
// 16-bit B-operand K index (same as A): v<4 -> K base 0, v>=4 -> K base 16;
// +8 per half-wave; VGPR v holds a pair of consecutive K.
__device__ __forceinline__ int kmapB(int v, int h) {
    return ((v & 4) << 2) + (h << 3) + ((v & 3) << 1);
}

__global__ __launch_bounds__(128)
void attn_fused_kernel(const float* __restrict__ Q,
                       const float* __restrict__ K,
                       const float* __restrict__ V,
                       const int*   __restrict__ M,
                       float* __restrict__ Out,
                       float* __restrict__ Pattn) {
    // ~131KB padded score tile (16 x 2052) + reductions + PV partials.
    __shared__ alignas(16) float sc[16][LDP];
    __shared__ float red[16][8];
    __shared__ float rowmax[16];
    __shared__ float rinv[16];
    __shared__ float pout[4][16][DIM];

    const int tid  = threadIdx.x;
    const int lane = tid & 31;
    const int wave = tid >> 5;
    const int half = lane >> 4;
    const int ln   = lane & 15;

    const int bq = blockIdx.x;          // BATCH * (SEQ/16) = 2048 blocks
    const int b  = bq >> 7;             // SEQ/16 = 128 query tiles per batch
    const int qt = bq & 127;
    const int q0 = qt << 4;

    const float* Qb = Q + ((size_t)(b * SEQ + q0)) * DIM;
    const float* Kb = K + ((size_t)b) * SEQ * DIM;
    const float* Vb = V + ((size_t)b) * SEQ * DIM;
    const int*   Mb = M + b * SEQ;

    // ---- Load Q tile into WMMA A-operand registers (f32 -> f16) ----------
    v16h a0, a1;                        // d = 0..31 and d = 32..63
    #pragma unroll
    for (int v = 0; v < 8; ++v) {
        const int kk = kmapA(v, half);
        const float* p = Qb + ln * DIM + kk;
        a0[2 * v]     = (_Float16)p[0];
        a0[2 * v + 1] = (_Float16)p[1];
        a1[2 * v]     = (_Float16)p[32];
        a1[2 * v + 1] = (_Float16)p[33];
    }

    // ---- Stage 1: scores = QK^T * scale, mask, into LDS ------------------
    for (int kt = wave; kt < SEQ / 16; kt += 4) {
        const int keyrow = (kt << 4) + ln;       // this lane's key (N column)
        // Prefetch this wave's next key tile (gfx1250 global_prefetch_b8).
        if (kt + 4 < SEQ / 16)
            __builtin_prefetch(Kb + (size_t)(keyrow + 64) * DIM, 0, 1);
        v16h b0, b1;
        #pragma unroll
        for (int v = 0; v < 8; ++v) {
            const int kk = kmapB(v, half);
            const float* p = Kb + (size_t)keyrow * DIM + kk;
            b0[2 * v]     = (_Float16)p[0];
            b0[2 * v + 1] = (_Float16)p[1];
            b1[2 * v]     = (_Float16)p[32];
            b1[2 * v + 1] = (_Float16)p[33];
        }
        v8f c = {};
        c = __builtin_amdgcn_wmma_f32_16x16x32_f16(false, a0, false, b0,
                                                   (short)0, c, false, false);
        c = __builtin_amdgcn_wmma_f32_16x16x32_f16(false, a1, false, b1,
                                                   (short)0, c, false, false);
        const int  col    = (kt << 4) + ln;
        const bool masked = (Mb[col] == 0);
        #pragma unroll
        for (int r = 0; r < 8; ++r) {
            float sv = c[r] * SCALE;
            if (masked) sv = MASK_NEG;
            sc[(half << 3) + r][col] = sv;       // row m = half*8 + r
        }
    }
    __syncthreads();

    // ---- Stage 2: softmax over each of the 16 rows -----------------------
    // row = tid&15, seg = tid>>4: 8 threads per row, float4 stride 8 ->
    // the 16 lanes of each half-wave hit all 64 LDS banks exactly once.
    const int srow = tid & 15;
    const int seg  = tid >> 4;

    float pmax = -3.4e38f;
    for (int i = 0; i < (SEQ / 4) / 8; ++i) {
        const int c = (seg + (i << 3)) << 2;
        const f32x4 vv = *reinterpret_cast<const f32x4*>(&sc[srow][c]);
        pmax = fmaxf(pmax, fmaxf(fmaxf(vv.x, vv.y), fmaxf(vv.z, vv.w)));
    }
    red[srow][seg] = pmax;
    __syncthreads();
    if (tid < 16) {
        float m = red[tid][0];
        #pragma unroll
        for (int i = 1; i < 8; ++i) m = fmaxf(m, red[tid][i]);
        rowmax[tid] = m;
    }
    __syncthreads();

    const float rm = rowmax[srow];
    float psum = 0.f;
    for (int i = 0; i < (SEQ / 4) / 8; ++i) {
        const int c = (seg + (i << 3)) << 2;
        f32x4 vv = *reinterpret_cast<const f32x4*>(&sc[srow][c]);
        vv.x = __expf(vv.x - rm);
        vv.y = __expf(vv.y - rm);
        vv.z = __expf(vv.z - rm);
        vv.w = __expf(vv.w - rm);
        *reinterpret_cast<f32x4*>(&sc[srow][c]) = vv;
        psum += vv.x + vv.y + vv.z + vv.w;
    }
    red[srow][seg] = psum;
    __syncthreads();
    if (tid < 16) {
        float s = 0.f;
        #pragma unroll
        for (int i = 0; i < 8; ++i) s += red[tid][i];
        rinv[tid] = 1.0f / s;
    }
    __syncthreads();

    // Normalize in LDS and stream p_attn to HBM with NON-TEMPORAL vec4
    // stores (written once, never re-read; keep K/V resident in the 192MB L2).
    float* prow = Pattn + ((size_t)(b * SEQ + q0)) * SEQ;
    for (int idx = tid; idx < (16 * SEQ) / 4; idx += 128) {
        const int linear = idx << 2;
        const int rr = linear >> 11;             // / SEQ
        const int cc = linear & (SEQ - 1);
        const float inv = rinv[rr];
        f32x4 pv = *reinterpret_cast<const f32x4*>(&sc[rr][cc]);
        pv.x *= inv; pv.y *= inv; pv.z *= inv; pv.w *= inv;
        *reinterpret_cast<f32x4*>(&sc[rr][cc]) = pv;
        __builtin_nontemporal_store(pv,
            reinterpret_cast<f32x4*>(prow + (size_t)rr * SEQ + cc));
    }
    __syncthreads();

    // ---- Stage 3: out = P . V  (split-K over the 4 waves) ----------------
    v8f acc0 = {}, acc1 = {}, acc2 = {}, acc3 = {};
    const int kstart = wave * (SEQ / 4);         // 512 keys per wave
    for (int ch = 0; ch < 16; ++ch) {            // 16 chunks of K=32
        const int kb0 = kstart + (ch << 5);
        v16h pa;                                  // A = probs 16x32 (from LDS)
        #pragma unroll
        for (int v = 0; v < 8; ++v) {
            const int kk = kmapA(v, half);
            pa[2 * v]     = (_Float16)sc[ln][kb0 + kk];
            pa[2 * v + 1] = (_Float16)sc[ln][kb0 + kk + 1];
        }
        #pragma unroll
        for (int nt = 0; nt < 4; ++nt) {         // D=64 -> 4 column tiles
            v16h vb;
            #pragma unroll
            for (int v = 0; v < 8; ++v) {
                const int kk = kmapB(v, half);
                const float* p = Vb + (size_t)(kb0 + kk) * DIM + (nt << 4) + ln;
                vb[2 * v]     = (_Float16)p[0];
                vb[2 * v + 1] = (_Float16)p[DIM];
            }
            switch (nt) {
            case 0: acc0 = __builtin_amdgcn_wmma_f32_16x16x32_f16(false, pa, false, vb, (short)0, acc0, false, false); break;
            case 1: acc1 = __builtin_amdgcn_wmma_f32_16x16x32_f16(false, pa, false, vb, (short)0, acc1, false, false); break;
            case 2: acc2 = __builtin_amdgcn_wmma_f32_16x16x32_f16(false, pa, false, vb, (short)0, acc2, false, false); break;
            case 3: acc3 = __builtin_amdgcn_wmma_f32_16x16x32_f16(false, pa, false, vb, (short)0, acc3, false, false); break;
            }
        }
    }
    #pragma unroll
    for (int r = 0; r < 8; ++r) {
        const int m = (half << 3) + r;
        pout[wave][m][ 0 + ln] = acc0[r];
        pout[wave][m][16 + ln] = acc1[r];
        pout[wave][m][32 + ln] = acc2[r];
        pout[wave][m][48 + ln] = acc3[r];
    }
    __syncthreads();

    // Cross-wave reduce and store the 16x64 output tile.
    float* ob = Out + ((size_t)(b * SEQ + q0)) * DIM;
    for (int e = tid; e < 16 * DIM; e += 128) {
        const int m  = e >> 6;
        const int dc = e & 63;
        const float s = pout[0][m][dc] + pout[1][m][dc] +
                        pout[2][m][dc] + pout[3][m][dc];
        __builtin_nontemporal_store(s, ob + (size_t)m * DIM + dc);
    }
}

extern "C" void kernel_launch(void* const* d_in, const int* in_sizes, int n_in,
                              void* d_out, int out_size, void* d_ws, size_t ws_size,
                              hipStream_t stream) {
    (void)in_sizes; (void)n_in; (void)d_ws; (void)ws_size; (void)out_size;
    const float* q = (const float*)d_in[0];
    const float* k = (const float*)d_in[1];
    const float* v = (const float*)d_in[2];
    const int*   m = (const int*)d_in[3];
    float* out   = (float*)d_out;                                   // B*S*D
    float* pattn = out + (size_t)BATCH * SEQ * DIM;                 // B*S*S
    const int blocks = BATCH * (SEQ / 16);                          // 2048
    attn_fused_kernel<<<blocks, 128, 0, stream>>>(q, k, v, m, out, pattn);
}
